// GMP_38345468018708
// MI455X (gfx1250) — matrix operand order repaired and checked by reading, not compiled, not run on hardware
//
#include <hip/hip_runtime.h>
#include <hip/hip_bf16.h>

// ---------------------------------------------------------------------------
// GNN message-passing layer for MI455X (gfx1250, wave32, WMMA bf16)
//   edge MLP (192->128->64) + LN + residual, segment-sum via f32 atomics,
//   node MLP (128->128->64) + LN + residual.
// All GEMMs on v_wmma_f32_16x16x32_bf16 (f32 accumulate).
// B-fragments batched into registers per k-step (deep DS pipeline);
// 4 tiles per wave to amortize weight preload.
// ---------------------------------------------------------------------------

#define N_NODES 100000
#define N_EDGES 1600000
#define ND 64
#define ED 64
#define HD 128
#define TPW 4            // tiles per wave

typedef __attribute__((ext_vector_type(16))) __bf16 v16bf;
typedef __attribute__((ext_vector_type(2)))  __bf16 bf16x2;
typedef __attribute__((ext_vector_type(8)))  float  v8f;

union Frag { v16bf v; uint4 q[2]; unsigned int u[8]; };
union Acc  { v8f  v; float f[8]; };

__device__ __forceinline__ unsigned int pack2(float a, float b) {
  bf16x2 t = { (__bf16)a, (__bf16)b };       // v_cvt_pk_bf16_f32
  return __builtin_bit_cast(unsigned int, t);
}
// convert 8 contiguous f32 (32B aligned) -> 4 packed bf16 dwords
__device__ __forceinline__ void cvt8(const float* __restrict__ p, unsigned int* dst) {
  float4 a = ((const float4*)p)[0];
  float4 b = ((const float4*)p)[1];
  dst[0] = pack2(a.x, a.y); dst[1] = pack2(a.z, a.w);
  dst[2] = pack2(b.x, b.y); dst[3] = pack2(b.z, b.w);
}

__global__ void gnn_zero_ws(float* __restrict__ p, int n) {
  int i = blockIdx.x * blockDim.x + threadIdx.x;
  int stride = gridDim.x * blockDim.x;
  for (; i < n; i += stride) p[i] = 0.0f;
}

// feature-segment pointer for edge input [x[src](64) | x[dst](64) | ea(64)]
__device__ __forceinline__ const float* eseg(const float* xs, const float* xd,
                                             const float* ear, int K) {
  if (K < 64)  return xs + K;
  if (K < 128) return xd + (K - 64);
  return ear + (K - 128);
}

// ---------------------------------------------------------------------------
// Edge kernel: 8 waves/block, TPW 16-edge tiles per wave.
// ---------------------------------------------------------------------------
__global__ __launch_bounds__(256) void gnn_edge_kernel(
    const float* __restrict__ x, const float* __restrict__ ea,
    const long long* __restrict__ ei,
    const float* __restrict__ eW1, const float* __restrict__ eb1,
    const float* __restrict__ eW2, const float* __restrict__ eb2,
    const float* __restrict__ eg,  const float* __restrict__ ebt,
    float* __restrict__ out_e, float* __restrict__ aggr)
{
  extern __shared__ char smem[];
  __bf16* wb1 = (__bf16*)smem;               // 8nt*6kb*32lane*16 = 24576 bf16
  __bf16* wb2 = wb1 + 8 * 6 * 512;           // 4nt*4kb*512       =  8192 bf16
  __bf16* stg = wb2 + 4 * 4 * 512;           // 8 waves * 16*128  = 16384 bf16

  const int tid = threadIdx.x;

  // --- preload weights into B-fragment LDS layout (bf16) -------------------
  // B frag (32x16 K x N): lane = 16*h + (n%16) holds K = K0+16h+i, i=0..15.
  for (int i = tid; i < 192 * 128; i += 256) {
    int k = i >> 7, n = i & 127;
    int nt = n >> 4, lnn = n & 15, kb = k >> 5, kr = k & 31;
    int hh = kr >> 4, ii = kr & 15;
    wb1[(((nt * 6 + kb) * 32) + (hh * 16 + lnn)) * 16 + ii] = (__bf16)eW1[i];
  }
  for (int i = tid; i < 128 * 64; i += 256) {
    int k = i >> 6, n = i & 63;
    int nt = n >> 4, lnn = n & 15, kb = k >> 5, kr = k & 31;
    int hh = kr >> 4, ii = kr & 15;
    wb2[(((nt * 4 + kb) * 32) + (hh * 16 + lnn)) * 16 + ii] = (__bf16)eW2[i];
  }
  __syncthreads();

  const int wave = tid >> 5;
  const int lane = tid & 31;
  const int ln = lane & 15;     // N-lane / A-row index
  const int h  = lane >> 4;     // lane half
  __bf16* st = stg + wave * 2048;            // per-wave 16x128 bf16 stage

  // per-lane bias/affine values (coalesced global reads, stay in registers)
  float eb1r[8], eb2r[4], egr[4], ebtr[4];
#pragma unroll
  for (int nt = 0; nt < 8; ++nt) eb1r[nt] = eb1[ln + 16 * nt];
#pragma unroll
  for (int nt = 0; nt < 4; ++nt) {
    eb2r[nt] = eb2[ln + 16 * nt];
    egr[nt]  = eg [ln + 16 * nt];
    ebtr[nt] = ebt[ln + 16 * nt];
  }

  const long long blockBase = (long long)blockIdx.x * (8 * TPW);

#pragma unroll 1
  for (int t = 0; t < TPW; ++t) {
    const long long tile  = blockBase + t * 8 + wave;   // waves stay contiguous
    const long long ebase = tile * 16;

    // A-operand gather pointers: this lane supplies row m = ln
    const long long eA = ebase + ln;
    const long long si = ei[eA];
    const long long di = ei[(long long)N_EDGES + eA];
    const float* xs   = x  + si * ND;
    const float* xd   = x  + di * ND;
    const float* eaRw = ea + eA * ED;

    // --- GEMM1: h[16x128] = edge_input[16x192] @ eW1 -----------------------
    Acc acc[8];
#pragma unroll
    for (int nt = 0; nt < 8; ++nt)
#pragma unroll
      for (int r = 0; r < 8; ++r) acc[nt].f[r] = 0.0f;

#pragma unroll
    for (int kb = 0; kb < 6; ++kb) {
      Frag a;
      const int K0 = kb * 32;
      cvt8(eseg(xs, xd, eaRw, K0 + 16 * h),     &a.u[0]);
      cvt8(eseg(xs, xd, eaRw, K0 + 16 * h + 8), &a.u[4]);
      // batch all 8 B-fragments first: one DS clause, staggered waits
      Frag bfr[8];
#pragma unroll
      for (int nt = 0; nt < 8; ++nt) {
        const uint4* bp = (const uint4*)(wb1 + (((nt * 6 + kb) * 32) + lane) * 16);
        bfr[nt].q[0] = bp[0]; bfr[nt].q[1] = bp[1];
      }
#pragma unroll
      for (int nt = 0; nt < 8; ++nt)
        acc[nt].v = __builtin_amdgcn_wmma_f32_16x16x32_bf16(
            false, a.v, false, bfr[nt].v, (short)0, acc[nt].v, false, false);
    }

    // --- ReLU+bias, stage h to LDS as bf16 (C-layout -> row-major) ---------
#pragma unroll
    for (int nt = 0; nt < 8; ++nt)
#pragma unroll
      for (int r = 0; r < 8; ++r) {
        const int n = ln + nt * 16;
        const int m = r + h * 8;
        st[m * 128 + n] = (__bf16)fmaxf(acc[nt].f[r] + eb1r[nt], 0.0f);
      }
    asm volatile("s_wait_dscnt 0x0" ::: "memory");  // wave-local LDS RAW

    // --- GEMM2: o[16x64] = h @ eW2 -----------------------------------------
    Acc acc2[4];
#pragma unroll
    for (int nt = 0; nt < 4; ++nt)
#pragma unroll
      for (int r = 0; r < 8; ++r) acc2[nt].f[r] = 0.0f;

#pragma unroll
    for (int kb = 0; kb < 4; ++kb) {
      Frag a;
      const uint4* p0 = (const uint4*)(st + ln * 128 + kb * 32 + 16 * h);
      const uint4* p1 = (const uint4*)(st + ln * 128 + kb * 32 + 16 * h + 8);
      a.q[0] = p0[0]; a.q[1] = p1[0];
      Frag bfr[4];
#pragma unroll
      for (int nt = 0; nt < 4; ++nt) {
        const uint4* bp = (const uint4*)(wb2 + (((nt * 4 + kb) * 32) + lane) * 16);
        bfr[nt].q[0] = bp[0]; bfr[nt].q[1] = bp[1];
      }
#pragma unroll
      for (int nt = 0; nt < 4; ++nt)
        acc2[nt].v = __builtin_amdgcn_wmma_f32_16x16x32_bf16(
            false, a.v, false, bfr[nt].v, (short)0, acc2[nt].v, false, false);
    }

    // --- bias + LayerNorm(64) + residual + store + scatter-add -------------
    float vals[4][8];
#pragma unroll
    for (int nt = 0; nt < 4; ++nt)
#pragma unroll
      for (int r = 0; r < 8; ++r) vals[nt][r] = acc2[nt].f[r] + eb2r[nt];

#pragma unroll
    for (int r = 0; r < 8; ++r) {
      float s  = vals[0][r] + vals[1][r] + vals[2][r] + vals[3][r];
      float ss = vals[0][r] * vals[0][r] + vals[1][r] * vals[1][r] +
                 vals[2][r] * vals[2][r] + vals[3][r] * vals[3][r];
#pragma unroll
      for (int msk = 1; msk < 16; msk <<= 1) {    // reduce within 16-lane half
        s  += __shfl_xor(s,  msk, 32);
        ss += __shfl_xor(ss, msk, 32);
      }
      const float mu   = s * (1.0f / 64.0f);
      const float var  = ss * (1.0f / 64.0f) - mu * mu;
      const float rinv = rsqrtf(var + 1e-5f);

      const long long eRow = ebase + r + 8 * h;
      const long long dIdx = ei[(long long)N_EDGES + eRow];
#pragma unroll
      for (int nt = 0; nt < 4; ++nt) {
        const int n = ln + nt * 16;
        float nv = (vals[nt][r] - mu) * rinv * egr[nt] + ebtr[nt];
        float ov = ea[eRow * ED + n] + nv;        // residual
        out_e[eRow * ED + n] = ov;
        atomicAdd(&aggr[dIdx * ED + n], ov);      // segment_sum over dst
      }
    }
  }
}

// ---------------------------------------------------------------------------
// Node kernel: 8 waves/block, TPW 16-node tiles per wave.
// ---------------------------------------------------------------------------
__global__ __launch_bounds__(256) void gnn_node_kernel(
    const float* __restrict__ x, const float* __restrict__ aggr,
    const float* __restrict__ nW1, const float* __restrict__ nb1,
    const float* __restrict__ nW2, const float* __restrict__ nb2,
    const float* __restrict__ ng,  const float* __restrict__ nbt,
    float* __restrict__ out_x)
{
  extern __shared__ char smem[];
  __bf16* wb1 = (__bf16*)smem;               // 8nt*4kb*512 = 16384 bf16
  __bf16* wb2 = wb1 + 8 * 4 * 512;           // 4nt*4kb*512 =  8192 bf16
  __bf16* stg = wb2 + 4 * 4 * 512;           // 8 * 2048    = 16384 bf16

  const int tid = threadIdx.x;
  for (int i = tid; i < 128 * 128; i += 256) {
    int k = i >> 7, n = i & 127;
    int nt = n >> 4, lnn = n & 15, kb = k >> 5, kr = k & 31;
    int hh = kr >> 4, ii = kr & 15;
    wb1[(((nt * 4 + kb) * 32) + (hh * 16 + lnn)) * 16 + ii] = (__bf16)nW1[i];
  }
  for (int i = tid; i < 128 * 64; i += 256) {
    int k = i >> 6, n = i & 63;
    int nt = n >> 4, lnn = n & 15, kb = k >> 5, kr = k & 31;
    int hh = kr >> 4, ii = kr & 15;
    wb2[(((nt * 4 + kb) * 32) + (hh * 16 + lnn)) * 16 + ii] = (__bf16)nW2[i];
  }
  __syncthreads();

  const int wave = tid >> 5;
  const int lane = tid & 31;
  const int ln = lane & 15;
  const int h  = lane >> 4;
  __bf16* st = stg + wave * 2048;

  float nb1r[8], nb2r[4], ngr[4], nbtr[4];
#pragma unroll
  for (int nt = 0; nt < 8; ++nt) nb1r[nt] = nb1[ln + 16 * nt];
#pragma unroll
  for (int nt = 0; nt < 4; ++nt) {
    nb2r[nt] = nb2[ln + 16 * nt];
    ngr[nt]  = ng [ln + 16 * nt];
    nbtr[nt] = nbt[ln + 16 * nt];
  }

  const long long blockBase = (long long)blockIdx.x * (8 * TPW);

#pragma unroll 1
  for (int t = 0; t < TPW; ++t) {
    const long long tile = blockBase + t * 8 + wave;
    if (tile >= N_NODES / 16) continue;
    const long long nbase = tile * 16;

    const long long nodeA = nbase + ln;           // A-row node for this lane
    const float* xsrc = x    + nodeA * ND;
    const float* asrc = aggr + nodeA * ED;

    // --- GEMM1: h[16x128] = [x|aggr][16x128] @ nW1 -------------------------
    Acc acc[8];
#pragma unroll
    for (int nt = 0; nt < 8; ++nt)
#pragma unroll
      for (int r = 0; r < 8; ++r) acc[nt].f[r] = 0.0f;

#pragma unroll
    for (int kb = 0; kb < 4; ++kb) {
      Frag a;
      const int K0 = kb * 32;
      const int k0 = K0 + 16 * h, k1 = K0 + 16 * h + 8;
      cvt8((k0 < 64) ? (xsrc + k0) : (asrc + k0 - 64), &a.u[0]);
      cvt8((k1 < 64) ? (xsrc + k1) : (asrc + k1 - 64), &a.u[4]);
      Frag bfr[8];
#pragma unroll
      for (int nt = 0; nt < 8; ++nt) {
        const uint4* bp = (const uint4*)(wb1 + (((nt * 4 + kb) * 32) + lane) * 16);
        bfr[nt].q[0] = bp[0]; bfr[nt].q[1] = bp[1];
      }
#pragma unroll
      for (int nt = 0; nt < 8; ++nt)
        acc[nt].v = __builtin_amdgcn_wmma_f32_16x16x32_bf16(
            false, a.v, false, bfr[nt].v, (short)0, acc[nt].v, false, false);
    }

#pragma unroll
    for (int nt = 0; nt < 8; ++nt)
#pragma unroll
      for (int r = 0; r < 8; ++r) {
        const int n = ln + nt * 16;
        const int m = r + h * 8;
        st[m * 128 + n] = (__bf16)fmaxf(acc[nt].f[r] + nb1r[nt], 0.0f);
      }
    asm volatile("s_wait_dscnt 0x0" ::: "memory");

    // --- GEMM2: o[16x64] = h @ nW2 -----------------------------------------
    Acc acc2[4];
#pragma unroll
    for (int nt = 0; nt < 4; ++nt)
#pragma unroll
      for (int r = 0; r < 8; ++r) acc2[nt].f[r] = 0.0f;

#pragma unroll
    for (int kb = 0; kb < 4; ++kb) {
      Frag a;
      const uint4* p0 = (const uint4*)(st + ln * 128 + kb * 32 + 16 * h);
      const uint4* p1 = (const uint4*)(st + ln * 128 + kb * 32 + 16 * h + 8);
      a.q[0] = p0[0]; a.q[1] = p1[0];
      Frag bfr[4];
#pragma unroll
      for (int nt = 0; nt < 4; ++nt) {
        const uint4* bp = (const uint4*)(wb2 + (((nt * 4 + kb) * 32) + lane) * 16);
        bfr[nt].q[0] = bp[0]; bfr[nt].q[1] = bp[1];
      }
#pragma unroll
      for (int nt = 0; nt < 4; ++nt)
        acc2[nt].v = __builtin_amdgcn_wmma_f32_16x16x32_bf16(
            false, a.v, false, bfr[nt].v, (short)0, acc2[nt].v, false, false);
    }

    float vals[4][8];
#pragma unroll
    for (int nt = 0; nt < 4; ++nt)
#pragma unroll
      for (int r = 0; r < 8; ++r) vals[nt][r] = acc2[nt].f[r] + nb2r[nt];

#pragma unroll
    for (int r = 0; r < 8; ++r) {
      float s  = vals[0][r] + vals[1][r] + vals[2][r] + vals[3][r];
      float ss = vals[0][r] * vals[0][r] + vals[1][r] * vals[1][r] +
                 vals[2][r] * vals[2][r] + vals[3][r] * vals[3][r];
#pragma unroll
      for (int msk = 1; msk < 16; msk <<= 1) {
        s  += __shfl_xor(s,  msk, 32);
        ss += __shfl_xor(ss, msk, 32);
      }
      const float mu   = s * (1.0f / 64.0f);
      const float var  = ss * (1.0f / 64.0f) - mu * mu;
      const float rinv = rsqrtf(var + 1e-5f);

      const long long nRow = nbase + r + 8 * h;
#pragma unroll
      for (int nt = 0; nt < 4; ++nt) {
        const int n = ln + nt * 16;
        float nv = (vals[nt][r] - mu) * rinv * ngr[nt] + nbtr[nt];
        out_x[nRow * ND + n] = x[nRow * ND + n] + nv; // residual
      }
    }
  }
}

// ---------------------------------------------------------------------------
extern "C" void kernel_launch(void* const* d_in, const int* in_sizes, int n_in,
                              void* d_out, int out_size, void* d_ws, size_t ws_size,
                              hipStream_t stream) {
  const float*     x   = (const float*)d_in[0];
  const float*     ea  = (const float*)d_in[1];
  const long long* ei  = (const long long*)d_in[2];   // int64 [2, E]
  const float* eW1 = (const float*)d_in[3];
  const float* eb1 = (const float*)d_in[4];
  const float* eW2 = (const float*)d_in[5];
  const float* eb2 = (const float*)d_in[6];
  const float* eg  = (const float*)d_in[7];
  const float* ebt = (const float*)d_in[8];
  const float* nW1 = (const float*)d_in[9];
  const float* nb1 = (const float*)d_in[10];
  const float* nW2 = (const float*)d_in[11];
  const float* nb2 = (const float*)d_in[12];
  const float* ng  = (const float*)d_in[13];
  const float* nbt = (const float*)d_in[14];

  float* out_x = (float*)d_out;                       // [N_NODES, ND]
  float* out_e = out_x + (size_t)N_NODES * ND;        // [N_EDGES, ED]
  float* aggr  = (float*)d_ws;                        // [N_NODES, ED] scratch

  // 1) zero the aggregation buffer (every call: deterministic)
  gnn_zero_ws<<<2048, 256, 0, stream>>>(aggr, N_NODES * ED);

  // 2) edge MLP + LN + residual + scatter-add
  const size_t smem_e = (size_t)(8 * 6 * 512 + 4 * 4 * 512 + 8 * 2048) * 2; // 98,304 B
  const int edge_tiles = N_EDGES / 16;                // 100,000
  gnn_edge_kernel<<<edge_tiles / (8 * TPW), 256, smem_e, stream>>>(
      x, ea, ei, eW1, eb1, eW2, eb2, eg, ebt, out_e, aggr);

  // 3) node MLP + LN + residual
  const size_t smem_n = (size_t)(8 * 4 * 512 + 4 * 4 * 512 + 8 * 2048) * 2; // 81,920 B
  const int node_tiles = N_NODES / 16;                // 6250
  gnn_node_kernel<<<(node_tiles + 8 * TPW - 1) / (8 * TPW), 256, smem_n, stream>>>(
      x, aggr, nW1, nb1, nW2, nb2, ng, nbt, out_x);
}
